// WordNet_55070070670242
// MI455X (gfx1250) — matrix operand order repaired and checked by reading, not compiled, not run on hardware
//
#include <hip/hip_runtime.h>
#include <hip/hip_bf16.h>
#include <math.h>

typedef __bf16 bf16;
typedef __attribute__((ext_vector_type(16))) __bf16 v16bf;
typedef __attribute__((ext_vector_type(8)))  __bf16 v8bf;
typedef __attribute__((ext_vector_type(4)))  __bf16 v4bf;
typedef __attribute__((ext_vector_type(8)))  float  v8f;

constexpr int NW = 50000, NS = 5000, E = 100000;
constexpr int HEADS = 6, C = 300, DS = 640, DW = 300, DGAT = 1800;

// order-preserving float->uint encoding of -INFINITY (0xFF800000 -> ~ -> 0x007FFFFF)
constexpr unsigned FENC_NEG_INF = 0x007FFFFFu;

// ---------- helpers ----------
__device__ __forceinline__ unsigned fenc(float f) {
    int i = __float_as_int(f);
    return (i >= 0) ? ((unsigned)i | 0x80000000u) : ~(unsigned)i;
}
__device__ __forceinline__ float fdec(unsigned u) {
    int i = (u & 0x80000000u) ? (int)(u & 0x7fffffffu) : (int)~u;
    return __int_as_float(i);
}

// ---------- elementwise kernels ----------
__global__ void cvt_bf16_kernel(const float* __restrict__ in, bf16* __restrict__ out, size_t n) {
    for (size_t i = (size_t)blockIdx.x * blockDim.x + threadIdx.x; i < n;
         i += (size_t)gridDim.x * blockDim.x)
        out[i] = (bf16)in[i];
}
__global__ void fill_f32_kernel(float* __restrict__ p, float v, size_t n) {
    for (size_t i = (size_t)blockIdx.x * blockDim.x + threadIdx.x; i < n;
         i += (size_t)gridDim.x * blockDim.x)
        p[i] = v;
}
__global__ void fill_u32_kernel(unsigned* __restrict__ p, unsigned v, size_t n) {
    for (size_t i = (size_t)blockIdx.x * blockDim.x + threadIdx.x; i < n;
         i += (size_t)gridDim.x * blockDim.x)
        p[i] = v;
}

// ---------- bf16 WMMA GEMM: C[M,N] = act(A[M,K] @ B[K,N] + bias + resid) ----------
// 256 threads = 8 waves arranged 4(M) x 2(N); each wave owns a 32x32 C tile
// (2x2 accumulators of 16x16), block tile 128x64, K stepped by 32 through LDS.
// sA is [m][k] row-major (64B rows); B is staged TRANSPOSED as sBt[n][k] so both
// A and B fragments are contiguous 16B runs per lane -> ds_load_b128 pairs.
__global__ void gemm_wmma_bf16(const bf16* __restrict__ A, const bf16* __restrict__ B,
                               const float* __restrict__ bias, const float* __restrict__ resid,
                               float* __restrict__ Cf, bf16* __restrict__ Cbf,
                               int M, int N, int K, int act) {
    __shared__ bf16 sA[128 * 32];  // 8 KB
    __shared__ bf16 sBt[64 * 32];  // 4 KB
    const int lane = threadIdx.x & 31;
    const int wave = threadIdx.x >> 5;  // 0..7
    const int half = lane >> 4;         // 0 or 1
    const int l16  = lane & 15;
    const int wm = wave >> 1;           // 0..3 -> M offset wm*32
    const int wn = wave & 1;            // 0..1 -> N offset wn*32
    const int mBase = blockIdx.x * 128;
    const int nBase = blockIdx.y * 64;

    v8f acc[2][2] = {};

    for (int k0 = 0; k0 < K; k0 += 32) {
        // ---- stage A tile 128x32 : one v4bf (8B) per task, coalesced along K
        for (int t = threadIdx.x; t < 128 * 8; t += 256) {
            int mm = t >> 3;        // 0..127
            int kv = (t & 7) * 4;   // 0,4,...,28
            int gm = mBase + mm, gk = k0 + kv;
            v4bf v = {};
            if (gm < M) {
                if (gk + 4 <= K) {
                    v = *(const v4bf*)(A + (size_t)gm * K + gk);
                } else {
                    for (int u = 0; u < 4; ++u)
                        v[u] = (gk + u < K) ? A[(size_t)gm * K + gk + u] : (bf16)0.0f;
                }
            }
            *(v4bf*)&sA[mm * 32 + kv] = v;
        }
        // ---- stage B tile 32x64 transposed: v4bf coalesced read along N,
        //      scalar transposed writes into sBt[n][k]
        for (int t = threadIdx.x; t < 32 * 16; t += 256) {
            int kk = t >> 4;         // 0..31
            int nv = (t & 15) * 4;   // 0,4,...,60
            int gk = k0 + kk, gn = nBase + nv;
            v4bf v = {};
            if (gk < K) {
                if (gn + 4 <= N) {
                    v = *(const v4bf*)(B + (size_t)gk * N + gn);
                } else {
                    for (int u = 0; u < 4; ++u)
                        v[u] = (gn + u < N) ? B[(size_t)gk * N + gn + u] : (bf16)0.0f;
                }
            }
            for (int u = 0; u < 4; ++u)
                sBt[(nv + u) * 32 + kk] = v[u];
        }
        // prefetch next K slab of A while we compute on this one
        if (k0 + 32 < K) {
            int gm = mBase + (int)(threadIdx.x >> 1);
            if (gm < M)
                __builtin_prefetch(A + (size_t)gm * K + k0 + 32 + (threadIdx.x & 1) * 16, 0, 3);
        }
        __syncthreads();

        // ---- fragments (ds_load_b128 pairs, contiguous per lane)
        // A frag: elems 0..7 -> k = half*8+i ; elems 8..15 -> k = 16+half*8+(i-8)
        v16bf afr[2], bfr[2];
#pragma unroll
        for (int i = 0; i < 2; ++i) {
            const bf16* r = &sA[(wm * 32 + i * 16 + l16) * 32];
            v8bf lo = *(const v8bf*)(r + half * 8);
            v8bf hi = *(const v8bf*)(r + 16 + half * 8);
            afr[i] = __builtin_shufflevector(lo, hi, 0, 1, 2, 3, 4, 5, 6, 7,
                                             8, 9, 10, 11, 12, 13, 14, 15);
        }
        // B frag: col = l16, elems 0..7 -> k = half*8+i ; elems 8..15 -> k = 16+half*8+(i-8)
#pragma unroll
        for (int j = 0; j < 2; ++j) {
            const bf16* r = &sBt[(wn * 32 + j * 16 + l16) * 32 + half * 8];
            v8bf lo = *(const v8bf*)(r);
            v8bf hi = *(const v8bf*)(r + 16);
            bfr[j] = __builtin_shufflevector(lo, hi, 0, 1, 2, 3, 4, 5, 6, 7,
                                             8, 9, 10, 11, 12, 13, 14, 15);
        }
#pragma unroll
        for (int i = 0; i < 2; ++i)
#pragma unroll
            for (int j = 0; j < 2; ++j)
                acc[i][j] = __builtin_amdgcn_wmma_f32_16x16x32_bf16(
                    false, afr[i], false, bfr[j], (short)0, acc[i][j], false, false);
        __syncthreads();
    }

    // ---- epilogue: C 16x16 f32 layout: acc[..][..][r] -> row = r + 8*half, col = l16
#pragma unroll
    for (int j = 0; j < 2; ++j) {
        int col = nBase + wn * 32 + j * 16 + l16;
        if (col >= N) continue;
        const float bv = bias ? bias[col] : 0.0f;
#pragma unroll
        for (int i = 0; i < 2; ++i) {
#pragma unroll
            for (int r = 0; r < 8; ++r) {
                int row = mBase + wm * 32 + i * 16 + r + half * 8;
                if (row < M) {
                    float v = acc[i][j][r] + bv;
                    if (resid) v += resid[(size_t)row * N + col];
                    if (act == 1) v = fmaxf(v, 0.0f);
                    size_t idx = (size_t)row * N + col;
                    if (Cf)  Cf[idx] = v;
                    if (Cbf) Cbf[idx] = (bf16)v;
                }
            }
        }
    }
}

// ---------- attention-logit kernels ----------
__global__ void a_s_kernel(const float* __restrict__ xs, const float* __restrict__ a_src,
                           float* __restrict__ a_s) {
    int t = blockIdx.x * blockDim.x + threadIdx.x;
    if (t >= NS * HEADS) return;
    int n = t / HEADS, h = t % HEADS;
    const float* x = xs + (size_t)n * DGAT + h * C;
    const float* a = a_src + h * C;
    float s = 0.f;
    for (int c = 0; c < C; ++c) s += x[c] * a[c];
    a_s[t] = s;
}
// v_dst[k,h] = sum_c W_dst[k, h*C+c] * a_dst[h,c]   (folds the 54 GF xd GEMM away)
__global__ void vdst_kernel(const float* __restrict__ W_dst, const float* __restrict__ a_dst,
                            float* __restrict__ v) {
    int t = blockIdx.x * blockDim.x + threadIdx.x;
    if (t >= DW * HEADS) return;
    int k = t / HEADS, h = t % HEADS;
    const float* w = W_dst + (size_t)k * DGAT + h * C;
    const float* a = a_dst + h * C;
    float s = 0.f;
    for (int c = 0; c < C; ++c) s += w[c] * a[c];
    v[t] = s;
}
__global__ void ad_kernel(const float* __restrict__ Hw, const float* __restrict__ v,
                          float* __restrict__ a_d) {
    int t = blockIdx.x * blockDim.x + threadIdx.x;
    if (t >= NW * HEADS) return;
    int n = t / HEADS, h = t % HEADS;
    const float* x = Hw + (size_t)n * DW;
    float s = 0.f;
    for (int k = 0; k < DW; ++k) s += x[k] * v[k * HEADS + h];
    a_d[t] = s;
}

// ---------- edge softmax ----------
__global__ void edge_score_kernel(const int* __restrict__ src, const int* __restrict__ dst,
                                  const float* __restrict__ a_s, const float* __restrict__ a_d,
                                  float* __restrict__ escore, unsigned* __restrict__ emax_u) {
    int t = blockIdx.x * blockDim.x + threadIdx.x;
    if (t >= E * HEADS) return;
    int e = t / HEADS, h = t % HEADS;
    float x = a_s[src[e] * HEADS + h] + a_d[dst[e] * HEADS + h];
    float sc = x > 0.f ? x : 0.2f * x;  // leaky_relu(0.2)
    escore[t] = sc;
    atomicMax(&emax_u[dst[e] * HEADS + h], fenc(sc));
}
__global__ void edge_exp_kernel(const int* __restrict__ dst, float* __restrict__ escore,
                                const unsigned* __restrict__ emax_u, float* __restrict__ denom) {
    int t = blockIdx.x * blockDim.x + threadIdx.x;
    if (t >= E * HEADS) return;
    int e = t / HEADS, h = t % HEADS;
    float ex = expf(escore[t] - fdec(emax_u[dst[e] * HEADS + h]));
    escore[t] = ex;  // in place: escore now holds exp()
    atomicAdd(&denom[dst[e] * HEADS + h], ex);
}
// one 256-thread block per edge: Uw[dst] += xs[src] * alpha  (coalesced over 1800)
__global__ void scatter_kernel(const int* __restrict__ src, const int* __restrict__ dst,
                               const float* __restrict__ xs, const float* __restrict__ exv,
                               const float* __restrict__ denom, float* __restrict__ Uw) {
    int e = blockIdx.x;
    int s = src[e], d = dst[e];
    __shared__ float alpha[HEADS];
    if (threadIdx.x < HEADS)
        alpha[threadIdx.x] = exv[e * HEADS + threadIdx.x] / denom[d * HEADS + threadIdx.x];
    __syncthreads();
    const float* xr = xs + (size_t)s * DGAT;
    float* ur = Uw + (size_t)d * DGAT;
    for (int j = threadIdx.x; j < DGAT; j += 256)
        atomicAdd(ur + j, xr[j] * alpha[j / C]);
}

// Uw_bf = bf16(elu(Uw + gat_bias))
__global__ void uw_finalize_kernel(const float* __restrict__ Uw, const float* __restrict__ bias,
                                   bf16* __restrict__ Uw_bf, size_t n) {
    for (size_t i = (size_t)blockIdx.x * blockDim.x + threadIdx.x; i < n;
         i += (size_t)gridDim.x * blockDim.x) {
        int col = (int)(i % DGAT);
        float v = Uw[i] + bias[col];
        v = v > 0.f ? v : expm1f(v);
        Uw_bf[i] = (bf16)v;
    }
}

// ---------- wave-per-row LayerNorm (wave32) ----------
__global__ void ln_kernel(const float* __restrict__ z, const float* __restrict__ g,
                          const float* __restrict__ b, float* __restrict__ out, int M, int N) {
    int wid = (int)(((size_t)blockIdx.x * blockDim.x + threadIdx.x) >> 5);
    int lane = threadIdx.x & 31;
    if (wid >= M) return;
    const float* row = z + (size_t)wid * N;
    float s = 0.f, s2 = 0.f;
    for (int j = lane; j < N; j += 32) {
        float v = row[j];
        s += v; s2 += v * v;
    }
#pragma unroll
    for (int o = 16; o > 0; o >>= 1) {
        s  += __shfl_down(s, o, 32);
        s2 += __shfl_down(s2, o, 32);
    }
    s  = __shfl(s, 0, 32);
    s2 = __shfl(s2, 0, 32);
    float mu = s / N;
    float var = s2 / N - mu * mu;
    float inv = rsqrtf(var + 1e-5f);
    float* orow = out + (size_t)wid * N;
    for (int j = lane; j < N; j += 32)
        orow[j] = g[j] * (row[j] - mu) * inv + b[j];
}

// ---------- driver ----------
extern "C" void kernel_launch(void* const* d_in, const int* in_sizes, int n_in,
                              void* d_out, int out_size, void* d_ws, size_t ws_size,
                              hipStream_t stream) {
    const float* Hw     = (const float*)d_in[0];
    const float* Hs     = (const float*)d_in[1];
    const int*   s2w    = (const int*)d_in[2];
    const float* W_src  = (const float*)d_in[3];
    const float* W_dst  = (const float*)d_in[4];
    const float* a_src  = (const float*)d_in[5];
    const float* a_dst  = (const float*)d_in[6];
    const float* gbias  = (const float*)d_in[7];
    const float* W1     = (const float*)d_in[8];
    const float* b1     = (const float*)d_in[9];
    const float* W2     = (const float*)d_in[10];
    const float* b2     = (const float*)d_in[11];
    const float* ln_g   = (const float*)d_in[12];
    const float* ln_b   = (const float*)d_in[13];
    const int* srcE = s2w;
    const int* dstE = s2w + E;
    float* out = (float*)d_out;

    // workspace carve-out (256B aligned)
    char* ws = (char*)d_ws;
    size_t off = 0;
    auto alloc = [&](size_t bytes) -> char* {
        char* p = ws + off;
        off += (bytes + 255) & ~(size_t)255;
        return p;
    };
    bf16*     Hs_bf   = (bf16*)alloc((size_t)NS * DS * 2);
    bf16*     Wsrc_bf = (bf16*)alloc((size_t)DS * DGAT * 2);
    bf16*     W1_bf   = (bf16*)alloc((size_t)DGAT * DW * 2);
    bf16*     W2_bf   = (bf16*)alloc((size_t)DW * DW * 2);
    float*    xs      = (float*)alloc((size_t)NS * DGAT * 4);
    float*    a_s     = (float*)alloc((size_t)NS * HEADS * 4);
    float*    vdst    = (float*)alloc((size_t)DW * HEADS * 4);
    float*    a_d     = (float*)alloc((size_t)NW * HEADS * 4);
    unsigned* emax_u  = (unsigned*)alloc((size_t)NW * HEADS * 4);
    float*    denom   = (float*)alloc((size_t)NW * HEADS * 4);
    float*    escore  = (float*)alloc((size_t)E * HEADS * 4);
    float*    Uw      = (float*)alloc((size_t)NW * DGAT * 4);
    bf16*     Uw_bf   = (bf16*)alloc((size_t)NW * DGAT * 2);
    bf16*     h_bf    = (bf16*)alloc((size_t)NW * DW * 2);
    float*    z       = (float*)alloc((size_t)NW * DW * 4);

    auto cdiv = [](long a, long b) { return (int)((a + b - 1) / b); };

    // bf16 conversions of GEMM operands
    cvt_bf16_kernel<<<cdiv((long)NS * DS, 256), 256, 0, stream>>>(Hs, Hs_bf, (size_t)NS * DS);
    cvt_bf16_kernel<<<cdiv((long)DS * DGAT, 256), 256, 0, stream>>>(W_src, Wsrc_bf, (size_t)DS * DGAT);
    cvt_bf16_kernel<<<cdiv((long)DGAT * DW, 256), 256, 0, stream>>>(W1, W1_bf, (size_t)DGAT * DW);
    cvt_bf16_kernel<<<cdiv((long)DW * DW, 256), 256, 0, stream>>>(W2, W2_bf, (size_t)DW * DW);

    // init accumulators (workspace is poisoned by harness)
    fill_f32_kernel<<<8192, 256, 0, stream>>>(Uw, 0.f, (size_t)NW * DGAT);
    fill_u32_kernel<<<cdiv((long)NW * HEADS, 256), 256, 0, stream>>>(emax_u, FENC_NEG_INF,
                                                                     (size_t)NW * HEADS);
    fill_f32_kernel<<<cdiv((long)NW * HEADS, 256), 256, 0, stream>>>(denom, 0.f,
                                                                     (size_t)NW * HEADS);

    // GEMM1: xs = Hs @ W_src   [5000,640]x[640,1800]
    {
        dim3 g(cdiv(NS, 128), cdiv(DGAT, 64));
        gemm_wmma_bf16<<<g, 256, 0, stream>>>(Hs_bf, Wsrc_bf, nullptr, nullptr, xs, nullptr,
                                              NS, DGAT, DS, 0);
    }
    // attention logits
    a_s_kernel<<<cdiv((long)NS * HEADS, 256), 256, 0, stream>>>(xs, a_src, a_s);
    vdst_kernel<<<cdiv((long)DW * HEADS, 256), 256, 0, stream>>>(W_dst, a_dst, vdst);
    ad_kernel<<<cdiv((long)NW * HEADS, 256), 256, 0, stream>>>(Hw, vdst, a_d);

    // edge softmax + scatter
    edge_score_kernel<<<cdiv((long)E * HEADS, 256), 256, 0, stream>>>(srcE, dstE, a_s, a_d,
                                                                      escore, emax_u);
    edge_exp_kernel<<<cdiv((long)E * HEADS, 256), 256, 0, stream>>>(dstE, escore, emax_u, denom);
    scatter_kernel<<<E, 256, 0, stream>>>(srcE, dstE, xs, escore, denom, Uw);

    // elu + bias, downcast for next GEMM
    uw_finalize_kernel<<<8192, 256, 0, stream>>>(Uw, gbias, Uw_bf, (size_t)NW * DGAT);

    // GEMM2: h = relu(Uw @ W1 + b1)  [50000,1800]x[1800,300] -> bf16
    {
        dim3 g(cdiv(NW, 128), cdiv(DW, 64));
        gemm_wmma_bf16<<<g, 256, 0, stream>>>(Uw_bf, W1_bf, b1, nullptr, nullptr, h_bf,
                                              NW, DW, DGAT, 1);
    }
    // GEMM3: z = Hw + h @ W2 + b2   [50000,300]x[300,300] -> f32
    {
        dim3 g(cdiv(NW, 128), cdiv(DW, 64));
        gemm_wmma_bf16<<<g, 256, 0, stream>>>(h_bf, W2_bf, b2, Hw, z, nullptr,
                                              NW, DW, DW, 0);
    }
    // LayerNorm -> output
    ln_kernel<<<cdiv(NW, 8), 256, 0, stream>>>(z, ln_g, ln_b, out, NW, DW);
}
